// LSSViewTransform_60670708023399
// MI455X (gfx1250) — matrix operand order repaired and checked by reading, not compiled, not run on hardware
//
#include <hip/hip_runtime.h>
#include <hip/hip_bf16.h>

typedef __attribute__((ext_vector_type(16))) _Float16 v16h;
typedef __attribute__((ext_vector_type(8)))  _Float16 v8h;
typedef __attribute__((ext_vector_type(8)))  float    v8f;

// ---- problem constants (match reference) ----
#define NB 2
#define NCAM 6
#define NC 64
#define FH 16
#define FW 44
#define ND 64
#define NPIX (FH*FW)            // 704
#define NIMG (NB*NCAM)          // 12
#define BEV_H 128
#define BEV_W 128
#define BEV_CELLS (BEV_H*BEV_W) // 16384
#define XMINF (-51.2f)
#define YMINF (-51.2f)
#define XRANGE 102.4f
#define YRANGE 102.4f
#define BN_EPSF 1e-5f
#define NPTS (NB*NCAM*ND*FH*FW) // 540672
#define W1P_HALVES (18*4*32*16) // 36864 packed conv1 weight halves
#define W2P_HALVES (2*4*32*16)  // 4096  packed conv2 weight halves

// ---------------------------------------------------------------------------
// Kernel 0: zero the BEV output (harness poisons d_out; scatter needs zeros)
// ---------------------------------------------------------------------------
__global__ void zero_out_kernel(float* __restrict__ out, int n4) {
  int i = blockIdx.x * blockDim.x + threadIdx.x;
  if (i < n4) ((float4*)out)[i] = make_float4(0.f, 0.f, 0.f, 0.f);
}

// ---------------------------------------------------------------------------
// Kernel P: pack conv weights (f32 -> f16) into exact per-lane WMMA B-fragment
// order, so the GEMM hot loops read them with two contiguous b128 loads.
// K-order for conv1: k = tap*64 + ci (tap = kh*3+kw), so within a 32-wide
// k-step the tap is fixed and ci is contiguous.
// Layout: [ks][ntile][lane][j0..15], half index = (((ks*4+nt)*32+lane)*16+j).
// ---------------------------------------------------------------------------
__global__ void pack_weights_kernel(const float* __restrict__ w1,
                                    const float* __restrict__ w2,
                                    _Float16* __restrict__ w1p,
                                    _Float16* __restrict__ w2p) {
  int i = blockIdx.x * blockDim.x + threadIdx.x;
  if (i < W1P_HALVES) {
    int j = i & 15, lane = (i >> 4) & 31, nt = (i >> 9) & 3, ks = i >> 11;
    int oc = nt * 16 + (lane & 15);
    int kB = ks * 32 + ((lane < 16) ? 0 : 8) + (j & 7) + ((j & 8) << 1);   // B layout: table order
    int tap = kB >> 6, ci = kB & 63;                 // k = tap*64 + ci
    w1p[i] = (_Float16)w1[oc * 576 + ci * 9 + tap];  // w1[oc][ci][kh][kw]
  }
  if (i < W2P_HALVES) {
    int j = i & 15, lane = (i >> 4) & 31, nt = (i >> 9) & 3, ks = i >> 11;
    int gd = nt * 16 + (lane & 15);
    int kB = ks * 32 + ((lane < 16) ? 0 : 8) + (j & 7) + ((j & 8) << 1);   // channel
    w2p[i] = (_Float16)w2[gd * NC + kB];             // w2[d][c]
  }
}

// ---------------------------------------------------------------------------
// Kernel 1: geometry -> per-(b,n,d,h,w) BEV flat index (or -1 if invalid)
// ---------------------------------------------------------------------------
__global__ void geom_kernel(const float* __restrict__ intr,
                            const float* __restrict__ extr,
                            int* __restrict__ idx_out) {
  int t = blockIdx.x * blockDim.x + threadIdx.x;
  if (t >= NPTS) return;
  int w = t % FW;
  int h = (t / FW) % FH;
  int d = (t / NPIX) % ND;
  int cam = (t / (NPIX * ND)) % NCAM;
  int b = t / (NPIX * ND * NCAM);

  float xs = (float)w * (703.0f / 43.0f);
  float ys = (float)h * (255.0f / 15.0f);
  float dv = 1.0f + (float)d * (59.0f / 63.0f);
  float px = xs * dv, py = ys * dv, pz = dv;

  const float* K = intr + (size_t)(b * NCAM + cam) * 9;
  float a00 = K[0], a01 = K[1], a02 = K[2];
  float a10 = K[3], a11 = K[4], a12 = K[5];
  float a20 = K[6], a21 = K[7], a22 = K[8];
  float c00 =  (a11 * a22 - a12 * a21);
  float c01 = -(a10 * a22 - a12 * a20);
  float c02 =  (a10 * a21 - a11 * a20);
  float inv = 1.0f / (a00 * c00 + a01 * c01 + a02 * c02);
  float i00 =  (a11 * a22 - a12 * a21) * inv;
  float i01 = -(a01 * a22 - a02 * a21) * inv;
  float i02 =  (a01 * a12 - a02 * a11) * inv;
  float i10 = -(a10 * a22 - a12 * a20) * inv;
  float i11 =  (a00 * a22 - a02 * a20) * inv;
  float i12 = -(a00 * a12 - a02 * a10) * inv;
  float i20 =  (a10 * a21 - a11 * a20) * inv;
  float i21 = -(a00 * a21 - a01 * a20) * inv;
  float i22 =  (a00 * a11 - a01 * a10) * inv;

  float cx = i00 * px + i01 * py + i02 * pz;
  float cy = i10 * px + i11 * py + i12 * pz;
  float cz = i20 * px + i21 * py + i22 * pz;

  const float* E = extr + (size_t)(b * NCAM + cam) * 16;
  float ex = E[0] * cx + E[1] * cy + E[2] * cz + E[3];
  float ey = E[4] * cx + E[5] * cy + E[6] * cz + E[7];

  int ix = (int)floorf((ex - XMINF) / XRANGE * (float)BEV_W);
  int iy = (int)floorf((ey - YMINF) / YRANGE * (float)BEV_H);
  bool valid = (ix >= 0) && (ix < BEV_W) && (iy >= 0) && (iy < BEV_H);
  idx_out[t] = valid ? (iy * BEV_W + ix) : -1;
}

// ---------------------------------------------------------------------------
// Kernel 2: conv1 3x3 SAME + bias + BN + ReLU via WMMA (im2col GEMM)
// grid = NIMG*2 (image x row-half), 256 threads (8 wave32).
// LDS: channel-LAST padded tile s_in[row(10)][col(46)][ci(64)] f16 (58,880 B)
// so an A fragment is two ds_load_b128 (ci contiguous).
// ks loop kept ROLLED (#pragma unroll 1) so the 8 B-fragment global loads per
// iteration track the induction pointer and cannot be mass-hoisted/spilled.
// Output h stored f16 pixel-major [img][pix][ch] for conv2's A fragments.
// ---------------------------------------------------------------------------
__global__ __launch_bounds__(256)
void conv1_bn_relu_kernel(const float* __restrict__ feat,
                          const _Float16* __restrict__ w1p,
                          const float* __restrict__ b1,
                          const float* __restrict__ gamma,
                          const float* __restrict__ beta,
                          const float* __restrict__ mean,
                          const float* __restrict__ var,
                          _Float16* __restrict__ h16) {
  __shared__ __align__(16) _Float16 s_in[10 * 46 * NC];

  const int img  = blockIdx.x >> 1;
  const int half = blockIdx.x & 1;
  const int h0   = half * 8;
  const int tid  = threadIdx.x;
  const int wave = tid >> 5;
  const int lane = tid & 31;

  __builtin_prefetch(w1p, 0, 0);   // global_prefetch_b8 on the weight stream

  // cooperative stage: rows h0-1..h0+8, cols -1..44, channel-last, f32->f16
  for (int i = tid; i < 10 * 46 * NC; i += 256) {
    int ci  = i & 63;
    int pos = i >> 6;              // r*46 + cc
    int r   = pos / 46;
    int cc  = pos - r * 46;
    int row = h0 + r - 1;
    int col = cc - 1;
    float v = 0.f;
    if (row >= 0 && row < FH && col >= 0 && col < FW)
      v = feat[((size_t)img * NC + ci) * NPIX + row * FW + col];
    s_in[i] = (_Float16)v;
  }
  __syncthreads();

  const int Am  = lane & 15;               // A: M = lane&15
  const int akb = (lane < 16) ? 0 : 8;     // A K-base per half-wave

  for (int mt = wave; mt < 22; mt += 8) {
    const int apix = mt * 16 + Am;         // local pixel 0..351
    const int arow = apix / FW;            // local row 0..7
    const int acol = apix - arow * FW;
    // per-lane base of the padded tile at (arow, acol), channel-last
    const _Float16* abase = &s_in[((arow * 46 + acol) << 6) + akb];
    v8f acc0 = {}, acc1 = {}, acc2 = {}, acc3 = {};

    const _Float16* wp = w1p + (lane << 4); // advances 2048 halves per ks

#pragma unroll 1
    for (int ks = 0; ks < 18; ++ks) {
      // tap = ks>>1 ; kh = tap/3 ; kw = tap%3  (cheap: tap <= 8)
      const int tap = ks >> 1;
      const int kh  = (tap * 11) >> 5;      // tap/3 for tap in [0,8]
      const int kw  = tap - kh * 3;
      // A fragment: halves j=0..7 -> ci=cib+j ; j=8..15 -> ci=cib+16+(j-8)
      const _Float16* ap = abase + (((kh * 46 + kw) << 6) + ((ks & 1) << 5));
      v8h alo = *(const v8h*)ap;            // ds_load_b128
      v8h ahi = *(const v8h*)(ap + 16);     // ds_load_b128
      v16h a = __builtin_shufflevector(alo, ahi, 0, 1, 2, 3, 4, 5, 6, 7,
                                       8, 9, 10, 11, 12, 13, 14, 15);
      // B fragments: 4 ntiles, contiguous packed halves (global, L2-hot)
      v8h b0l = *(const v8h*)(wp + 0 * 512);
      v8h b0h = *(const v8h*)(wp + 0 * 512 + 8);
      v8h b1l = *(const v8h*)(wp + 1 * 512);
      v8h b1h = *(const v8h*)(wp + 1 * 512 + 8);
      v8h b2l = *(const v8h*)(wp + 2 * 512);
      v8h b2h = *(const v8h*)(wp + 2 * 512 + 8);
      v8h b3l = *(const v8h*)(wp + 3 * 512);
      v8h b3h = *(const v8h*)(wp + 3 * 512 + 8);
      v16h bf0 = __builtin_shufflevector(b0l, b0h, 0, 1, 2, 3, 4, 5, 6, 7,
                                         8, 9, 10, 11, 12, 13, 14, 15);
      v16h bf1 = __builtin_shufflevector(b1l, b1h, 0, 1, 2, 3, 4, 5, 6, 7,
                                         8, 9, 10, 11, 12, 13, 14, 15);
      v16h bf2 = __builtin_shufflevector(b2l, b2h, 0, 1, 2, 3, 4, 5, 6, 7,
                                         8, 9, 10, 11, 12, 13, 14, 15);
      v16h bf3 = __builtin_shufflevector(b3l, b3h, 0, 1, 2, 3, 4, 5, 6, 7,
                                         8, 9, 10, 11, 12, 13, 14, 15);
      acc0 = __builtin_amdgcn_wmma_f32_16x16x32_f16(false, a, false, bf0,
                                                    (short)0, acc0, false, false);
      acc1 = __builtin_amdgcn_wmma_f32_16x16x32_f16(false, a, false, bf1,
                                                    (short)0, acc1, false, false);
      acc2 = __builtin_amdgcn_wmma_f32_16x16x32_f16(false, a, false, bf2,
                                                    (short)0, acc2, false, false);
      acc3 = __builtin_amdgcn_wmma_f32_16x16x32_f16(false, a, false, bf3,
                                                    (short)0, acc3, false, false);
      wp += 2048;                           // next ks block (4 nt * 512)
    }

    // epilogue: bias + BN + ReLU, store f16 pixel-major [img][pix][ch]
    const int nloc = lane & 15;
    const int pmb  = (lane < 16) ? 0 : 8;   // C/D: VGPR r -> M=r (+8 hi half)
#pragma unroll
    for (int nt = 0; nt < 4; ++nt) {
      const int oc = nt * 16 + nloc;
      const float sc = gamma[oc] * __frsqrt_rn(var[oc] + BN_EPSF);
      const float bb = beta[oc] - mean[oc] * sc;
      const float bi = b1[oc];
      const v8f acc = (nt == 0) ? acc0 : (nt == 1) ? acc1
                    : (nt == 2) ? acc2 : acc3;
#pragma unroll
      for (int r = 0; r < 8; ++r) {
        int pix = mt * 16 + r + pmb;        // local pixel
        float v = (acc[r] + bi) * sc + bb;
        v = v > 0.f ? v : 0.f;
        h16[((size_t)img * NPIX + h0 * FW + pix) * NC + oc] = (_Float16)v;
      }
    }
  }
}

// ---------------------------------------------------------------------------
// Kernel 3: conv2 1x1 (WMMA, K=64) + softmax over 64 depths + volume*scatter
// grid = NIMG*44 (16-pixel tiles), 128 threads (4 wave32, wave = depth tile).
// h16 is pixel-major, so A fragments are two global_load_b128 per k-step.
// ---------------------------------------------------------------------------
__global__ __launch_bounds__(128)
void conv2_softmax_scatter_kernel(const float* __restrict__ feat,
                                  const _Float16* __restrict__ h16,
                                  const _Float16* __restrict__ w2p,
                                  const float* __restrict__ b2,
                                  const int* __restrict__ idxbuf,
                                  float* __restrict__ out) {
  __shared__ float s_logits[ND * 16];   // [depth][pixel]
  __shared__ float s_feat[NC * 16];     // [channel][pixel]

  const int img   = blockIdx.x / 44;
  const int mtile = blockIdx.x % 44;
  const int gbase = mtile * 16;
  const int b     = img / NCAM;
  const int tid   = threadIdx.x;
  const int nt    = tid >> 5;
  const int lane  = tid & 31;

  for (int i = tid; i < NC * 16; i += 128) {
    int c = i >> 4, p = i & 15;
    s_feat[i] = feat[((size_t)img * NC + c) * NPIX + gbase + p];
  }

  v8f acc = {};
  const int Am  = lane & 15;
  const int akb = (lane < 16) ? 0 : 8;
  const _Float16* hrow = h16 + ((size_t)img * NPIX + gbase + Am) * NC;
#pragma unroll
  for (int ks = 0; ks < 2; ++ks) {
    const _Float16* ap = hrow + ks * 32 + akb;
    v8h alo = *(const v8h*)ap;             // global_load_b128
    v8h ahi = *(const v8h*)(ap + 16);      // global_load_b128
    v16h a = __builtin_shufflevector(alo, ahi, 0, 1, 2, 3, 4, 5, 6, 7,
                                     8, 9, 10, 11, 12, 13, 14, 15);
    const _Float16* wp = w2p + (((ks * 4 + nt) * 32 + lane) << 4);
    v8h blo = *(const v8h*)wp;
    v8h bhi = *(const v8h*)(wp + 8);
    v16h bf = __builtin_shufflevector(blo, bhi, 0, 1, 2, 3, 4, 5, 6, 7,
                                      8, 9, 10, 11, 12, 13, 14, 15);
    acc = __builtin_amdgcn_wmma_f32_16x16x32_f16(false, a, false, bf,
                                                 (short)0, acc, false, false);
  }

  const int nloc = lane & 15;
  const float bi = b2[nt * 16 + nloc];
#pragma unroll
  for (int r = 0; r < 8; ++r) {
    int pm = r + ((lane < 16) ? 0 : 8);
    s_logits[(nt * 16 + nloc) * 16 + pm] = acc[r] + bi;
  }
  __syncthreads();

  if (tid < 16) {
    const int p = tid;
    float mx = -3.4e38f;
    for (int d = 0; d < ND; ++d) mx = fmaxf(mx, s_logits[d * 16 + p]);
    float sum = 0.f;
    for (int d = 0; d < ND; ++d) sum += __expf(s_logits[d * 16 + p] - mx);
    float rs = 1.0f / sum;
    for (int d = 0; d < ND; ++d)
      s_logits[d * 16 + p] = __expf(s_logits[d * 16 + p] - mx) * rs;
  }
  __syncthreads();

  // volume = x * depth_prob, scatter-add into BEV (atomics land in L2)
  for (int pair = tid; pair < 16 * ND; pair += 128) {
    const int p = pair & 15;
    const int d = pair >> 4;
    const float prob = s_logits[d * 16 + p];
    const int bidx = idxbuf[((size_t)img * ND + d) * NPIX + gbase + p];
    if (bidx < 0) continue;
    float* outc = out + (size_t)b * NC * BEV_CELLS + bidx;
#pragma unroll 4
    for (int c = 0; c < NC; ++c)
      atomicAdd(outc + (size_t)c * BEV_CELLS, s_feat[c * 16 + p] * prob);
  }
}

// ---------------------------------------------------------------------------
extern "C" void kernel_launch(void* const* d_in, const int* in_sizes, int n_in,
                              void* d_out, int out_size, void* d_ws, size_t ws_size,
                              hipStream_t stream) {
  (void)in_sizes; (void)n_in; (void)out_size; (void)ws_size;
  const float* feat = (const float*)d_in[0];
  const float* intr = (const float*)d_in[1];
  const float* extr = (const float*)d_in[2];
  const float* w1   = (const float*)d_in[3];
  const float* b1   = (const float*)d_in[4];
  const float* gam  = (const float*)d_in[5];
  const float* bet  = (const float*)d_in[6];
  const float* mean = (const float*)d_in[7];
  const float* var  = (const float*)d_in[8];
  const float* w2   = (const float*)d_in[9];
  const float* b2   = (const float*)d_in[10];
  float* out = (float*)d_out;

  char* ws = (char*)d_ws;
  int*      idx_ws = (int*)ws;                         ws += (size_t)NPTS * 4;
  _Float16* h16    = (_Float16*)ws;                    ws += (size_t)NIMG * NPIX * NC * 2;
  _Float16* w1p    = (_Float16*)ws;                    ws += (size_t)W1P_HALVES * 2;
  _Float16* w2p    = (_Float16*)ws;

  zero_out_kernel<<<(NB * NC * BEV_CELLS / 4 + 255) / 256, 256, 0, stream>>>(
      out, NB * NC * BEV_CELLS / 4);
  pack_weights_kernel<<<(W1P_HALVES + 255) / 256, 256, 0, stream>>>(
      w1, w2, w1p, w2p);
  geom_kernel<<<(NPTS + 255) / 256, 256, 0, stream>>>(intr, extr, idx_ws);
  conv1_bn_relu_kernel<<<NIMG * 2, 256, 0, stream>>>(
      feat, w1p, b1, gam, bet, mean, var, h16);
  conv2_softmax_scatter_kernel<<<NIMG * 44, 128, 0, stream>>>(
      feat, h16, w2p, b2, idx_ws, out);
}
